// VisionMambaQuant_19980187861555
// MI455X (gfx1250) — matrix operand order repaired and checked by reading, not compiled, not run on hardware
//
#include <hip/hip_runtime.h>
#include <hip/hip_bf16.h>

typedef __attribute__((ext_vector_type(16))) _Float16 v16h;
typedef __attribute__((ext_vector_type(8)))  _Float16 v8h;
typedef __attribute__((ext_vector_type(8)))  float    v8f;

#define D_MODEL 192
#define D_INNER 384
#define D_STATE 16
#define D_CONVK 4
#define DT_RANK 12
#define DEPTH 24
#define LSEQ 197
#define BATCH 8
#define NTOK (BATCH * LSEQ)                 // 1576 tokens
#define XPROJ_OUT (DT_RANK + 2 * D_STATE)   // 44

__device__ __forceinline__ float sigmoidf_(float x) { return 1.f / (1.f + __expf(-x)); }

// ---------------------------------------------------------------------------
// f32 -> f16 weight conversion (weights fit in L2; converted once per launch)
// ---------------------------------------------------------------------------
__global__ void k_f32_to_f16(const float* __restrict__ src, _Float16* __restrict__ dst, int n) {
  int i = blockIdx.x * 256 + threadIdx.x;
  if (i < n) dst[i] = (_Float16)src[i];
}

// ---------------------------------------------------------------------------
// Patch embed (im2col dot per output channel) + cls insert + pos_embed.
// Also zero-initializes the residual stream. grid (197, 8), block 192
// ---------------------------------------------------------------------------
__global__ void k_patch_embed(const float* __restrict__ x,  const float* __restrict__ pw,
                              const float* __restrict__ pb, const float* __restrict__ ct,
                              const float* __restrict__ pe, float* __restrict__ hidden,
                              float* __restrict__ residual) {
  int tok = blockIdx.x;      // 0..196
  int b   = blockIdx.y;
  int c   = threadIdx.x;     // 0..191
  float val;
  if (tok == 98) {
    val = ct[c];
  } else {
    int p  = (tok < 98) ? tok : tok - 1;
    int py = p / 14, px = p % 14;
    float acc = 0.f;
    for (int ci = 0; ci < 3; ++ci)
      for (int ky = 0; ky < 16; ++ky) {
        const float* xr = x  + (((size_t)(b * 3 + ci) * 224 + py * 16 + ky) * 224 + px * 16);
        const float* wr = pw + (((size_t)(c * 3 + ci) * 16 + ky) * 16);
#pragma unroll
        for (int kx = 0; kx < 16; ++kx) acc += xr[kx] * wr[kx];
      }
    val = acc + pb[c];
  }
  val += pe[tok * D_MODEL + c];
  size_t ix = ((size_t)b * LSEQ + tok) * D_MODEL + c;
  hidden[ix]   = val;
  residual[ix] = 0.f;
}

// ---------------------------------------------------------------------------
// residual = hidden + residual; xnorm_f16 = RMSNorm(residual) * norm_w
// grid NTOK, block 192
// ---------------------------------------------------------------------------
__global__ void k_prenorm(const float* __restrict__ hidden, float* __restrict__ residual,
                          const float* __restrict__ nw, _Float16* __restrict__ xnorm) {
  int t = blockIdx.x;
  int c = threadIdx.x;
  __shared__ float red[192];
  __shared__ float rs_s;
  size_t ix = (size_t)t * D_MODEL + c;
  float hv = hidden[ix] + residual[ix];
  residual[ix] = hv;
  red[c] = hv * hv;
  __syncthreads();
  if (c < 64) red[c] += red[c + 64] + red[c + 128];
  __syncthreads();
  if (c < 32) red[c] += red[c + 32];
  __syncthreads();
  if (c == 0) {
    float s = 0.f;
    for (int i = 0; i < 32; ++i) s += red[i];
    rs_s = rsqrtf(s / (float)D_MODEL + 1e-5f);
  }
  __syncthreads();
  xnorm[ix] = (_Float16)(hv * rs_s * nw[c]);
}

// ---------------------------------------------------------------------------
// WMMA fragment loaders (16-bit A 16x32 / B 32x16); contiguous k order, same for A and B.
// A: lane m=lane%16, g=lane/16, element e -> K = e + 16*g
//    => one contiguous 32B load at k0+16g.
// B: lane n=lane%16, g=lane/16, element e -> K = e + 16*g
//    => one contiguous 32B load at k0+16g.
// C/D: lane n=lane%16, reg r -> M = r + 8*(lane/16).
// ---------------------------------------------------------------------------
__device__ __forceinline__ v16h load_a_frag(const _Float16* __restrict__ arow, int k0, int g) {
  return *(const v16h*)(arow + k0 + g * 16);
}
__device__ __forceinline__ v16h load_b_frag(const _Float16* __restrict__ brow, int k0, int g) {
  return *(const v16h*)(brow + k0 + g * 16);
}

// ---------------------------------------------------------------------------
// 1-tile WMMA GEMM (used for x_proj, N=44): Y[M,Nvalid] = X[M,K] * W[Nvalid,K]^T
// block = 32 (one wave per 16x16 tile)
// ---------------------------------------------------------------------------
template <int K>
__global__ void k_gemm_f16(const _Float16* __restrict__ X, const _Float16* __restrict__ W,
                           float* __restrict__ Y, int M, int Nvalid, int ldY, int tilesN) {
  int tile = blockIdx.x;
  int tN = tile % tilesN;
  int tM = tile / tilesN;
  int lane = threadIdx.x;
  int mr = lane & 15;
  int g  = lane >> 4;

  int rowA = tM * 16 + mr; if (rowA >= M)      rowA = 0;
  int colB = tN * 16 + mr; if (colB >= Nvalid) colB = 0;

  const _Float16* arow = X + (size_t)rowA * K;
  const _Float16* brow = W + (size_t)colB * K;

  v8f acc = {};
#pragma unroll
  for (int k0 = 0; k0 < K; k0 += 32) {
    v16h a = load_a_frag(arow, k0, g);
    v16h b = load_b_frag(brow, k0, g);
    acc = __builtin_amdgcn_wmma_f32_16x16x32_f16(false, a, false, b, (short)0, acc, false, false);
  }

  int n = tN * 16 + mr;
  if (n < Nvalid) {
    int mbase = tM * 16 + g * 8;
#pragma unroll
    for (int r = 0; r < 8; ++r) {
      int m = mbase + r;
      if (m < M) Y[(size_t)m * ldY + n] = acc[r];
    }
  }
}

// ---------------------------------------------------------------------------
// 2x2 register-tiled WMMA GEMM (in_proj / out_proj): one wave computes a
// 32x32 output block (4 accumulators). Each A fragment is reused across 2
// N-tiles and each B fragment across 2 M-tiles: 32 B/lane of loads per WMMA
// vs 64 B/lane for the 1-tile kernel -> load-issue no longer the bottleneck.
// Requires N % 32 == 0. block = 32
// ---------------------------------------------------------------------------
template <int K>
__global__ void k_gemm_f16_2x2(const _Float16* __restrict__ X, const _Float16* __restrict__ W,
                               float* __restrict__ Y, int M, int N, int ldY, int tilesN2) {
  int tile = blockIdx.x;
  int tN = tile % tilesN2;       // 32-col block
  int tM = tile / tilesN2;       // 32-row block
  int lane = threadIdx.x;
  int mr = lane & 15;
  int g  = lane >> 4;

  int rowA0 = tM * 32 + mr;      if (rowA0 >= M) rowA0 = 0;
  int rowA1 = tM * 32 + 16 + mr; if (rowA1 >= M) rowA1 = 0;
  int colB0 = tN * 32 + mr;      // N % 32 == 0 -> always valid
  int colB1 = tN * 32 + 16 + mr;

  const _Float16* arow0 = X + (size_t)rowA0 * K;
  const _Float16* arow1 = X + (size_t)rowA1 * K;
  const _Float16* brow0 = W + (size_t)colB0 * K;
  const _Float16* brow1 = W + (size_t)colB1 * K;

  v8f acc00 = {}, acc01 = {}, acc10 = {}, acc11 = {};
#pragma unroll
  for (int k0 = 0; k0 < K; k0 += 32) {
    v16h a0 = load_a_frag(arow0, k0, g);
    v16h a1 = load_a_frag(arow1, k0, g);
    v16h b0 = load_b_frag(brow0, k0, g);
    v16h b1 = load_b_frag(brow1, k0, g);
    acc00 = __builtin_amdgcn_wmma_f32_16x16x32_f16(false, a0, false, b0, (short)0, acc00, false, false);
    acc01 = __builtin_amdgcn_wmma_f32_16x16x32_f16(false, a0, false, b1, (short)0, acc01, false, false);
    acc10 = __builtin_amdgcn_wmma_f32_16x16x32_f16(false, a1, false, b0, (short)0, acc10, false, false);
    acc11 = __builtin_amdgcn_wmma_f32_16x16x32_f16(false, a1, false, b1, (short)0, acc11, false, false);
  }

  int n0 = tN * 32 + mr;
  int n1 = tN * 32 + 16 + mr;
  int mb0 = tM * 32 + g * 8;
  int mb1 = tM * 32 + 16 + g * 8;
#pragma unroll
  for (int r = 0; r < 8; ++r) {
    int m0 = mb0 + r, m1 = mb1 + r;
    if (m0 < M) {
      Y[(size_t)m0 * ldY + n0] = acc00[r];
      Y[(size_t)m0 * ldY + n1] = acc01[r];
    }
    if (m1 < M) {
      Y[(size_t)m1 * ldY + n0] = acc10[r];
      Y[(size_t)m1 * ldY + n1] = acc11[r];
    }
  }
}

// ---------------------------------------------------------------------------
// Depthwise causal conv1d (k=4) + bias + SiLU; writes f32 (scan) and f16 (x_proj)
// ---------------------------------------------------------------------------
__global__ void k_conv_silu(const float* __restrict__ xz, const float* __restrict__ cw,
                            const float* __restrict__ cb, float* __restrict__ xc,
                            _Float16* __restrict__ xch) {
  int idx = blockIdx.x * 256 + threadIdx.x;
  if (idx >= NTOK * D_INNER) return;
  int d = idx % D_INNER;
  int t = idx / D_INNER;
  int b = t / LSEQ, l = t % LSEQ;
  float acc = cb[d];
#pragma unroll
  for (int k = 0; k < D_CONVK; ++k) {
    int ll = l + k - (D_CONVK - 1);
    if (ll >= 0) acc += cw[d * D_CONVK + k] * xz[((size_t)(b * LSEQ + ll)) * (2 * D_INNER) + d];
  }
  float s = acc * sigmoidf_(acc);
  xc[idx]  = s;
  xch[idx] = (_Float16)s;
}

// ---------------------------------------------------------------------------
// dt = softplus(dt_part @ dtw^T + dtb); K=12 so VALU dot
// ---------------------------------------------------------------------------
__global__ void k_dt_softplus(const float* __restrict__ xdbl, const float* __restrict__ dtw,
                              const float* __restrict__ dtb, float* __restrict__ dtf) {
  int idx = blockIdx.x * 256 + threadIdx.x;
  if (idx >= NTOK * D_INNER) return;
  int d = idx % D_INNER;
  int t = idx / D_INNER;
  const float* row = xdbl + (size_t)t * XPROJ_OUT;
  float acc = dtb[d];
#pragma unroll
  for (int r = 0; r < DT_RANK; ++r) acc += row[r] * dtw[d * DT_RANK + r];
  dtf[idx] = (acc > 20.f) ? acc : log1pf(__expf(acc));
}

// ---------------------------------------------------------------------------
// Selective scan + SiLU(z) gate fused. One thread per (b,d); h[16] in regs,
// B/C staged in LDS per step. grid (3, 8), block 128.
// ---------------------------------------------------------------------------
__global__ void k_scan_gate(const float* __restrict__ dtf, const float* __restrict__ xc,
                            const float* __restrict__ xdbl, const float* __restrict__ xz,
                            const float* __restrict__ Alog, const float* __restrict__ Dp,
                            _Float16* __restrict__ ygh) {
  int b = blockIdx.y;
  int d = blockIdx.x * 128 + threadIdx.x;   // 0..383
  float negA[D_STATE], h[D_STATE];
#pragma unroll
  for (int n = 0; n < D_STATE; ++n) { negA[n] = -__expf(Alog[d * D_STATE + n]); h[n] = 0.f; }
  float Dd = Dp[d];
  __shared__ float Bs[D_STATE], Cs[D_STATE];
  for (int l = 0; l < LSEQ; ++l) {
    __syncthreads();
    if (threadIdx.x < 32) {
      const float* r = xdbl + ((size_t)(b * LSEQ + l)) * XPROJ_OUT;
      if (threadIdx.x < 16) Bs[threadIdx.x] = r[DT_RANK + threadIdx.x];
      else                  Cs[threadIdx.x - 16] = r[DT_RANK + threadIdx.x];
    }
    __syncthreads();
    size_t ix = ((size_t)(b * LSEQ + l)) * D_INNER + d;
    float dt = dtf[ix], u = xc[ix];
    float du = dt * u, y = 0.f;
#pragma unroll
    for (int n = 0; n < D_STATE; ++n) {
      h[n] = __expf(dt * negA[n]) * h[n] + du * Bs[n];
      y += h[n] * Cs[n];
    }
    y += u * Dd;
    float z = xz[((size_t)(b * LSEQ + l)) * (2 * D_INNER) + D_INNER + d];
    ygh[ix] = (_Float16)(y * (z * sigmoidf_(z)));
  }
}

// ---------------------------------------------------------------------------
// Final: RMSNorm(hidden+residual) at cls token (98) + head GEMV. grid 8, blk 256
// ---------------------------------------------------------------------------
__global__ void k_head(const float* __restrict__ hidden, const float* __restrict__ residual,
                       const float* __restrict__ nfw, const float* __restrict__ hw,
                       const float* __restrict__ hb, float* __restrict__ out) {
  int b = blockIdx.x;
  int tid = threadIdx.x;
  __shared__ float v[D_MODEL];
  __shared__ float red[256];
  __shared__ float rs_s;
  float hv = 0.f;
  if (tid < D_MODEL) {
    size_t ix = ((size_t)(b * LSEQ + 98)) * D_MODEL + tid;
    hv = hidden[ix] + residual[ix];
  }
  red[tid] = hv * hv;
  __syncthreads();
  for (int s = 128; s > 0; s >>= 1) { if (tid < s) red[tid] += red[tid + s]; __syncthreads(); }
  if (tid == 0) rs_s = rsqrtf(red[0] / (float)D_MODEL + 1e-5f);
  __syncthreads();
  if (tid < D_MODEL) v[tid] = hv * rs_s * nfw[tid];
  __syncthreads();
  for (int o = tid; o < 1000; o += 256) {
    float acc = hb[o];
    const float* wr = hw + (size_t)o * D_MODEL;
#pragma unroll 4
    for (int c = 0; c < D_MODEL; ++c) acc += v[c] * wr[c];
    out[b * 1000 + o] = acc;
  }
}

// ---------------------------------------------------------------------------
extern "C" void kernel_launch(void* const* d_in, const int* in_sizes, int n_in,
                              void* d_out, int out_size, void* d_ws, size_t ws_size,
                              hipStream_t stream) {
  (void)in_sizes; (void)n_in; (void)out_size; (void)ws_size;
  const float* x    = (const float*)d_in[0];
  const float* pw   = (const float*)d_in[1];
  const float* pb   = (const float*)d_in[2];
  const float* ct   = (const float*)d_in[3];
  const float* pe   = (const float*)d_in[4];
  const float* nw   = (const float*)d_in[5];
  const float* inw  = (const float*)d_in[6];
  const float* cw   = (const float*)d_in[7];
  const float* cb   = (const float*)d_in[8];
  const float* xw   = (const float*)d_in[9];
  const float* dtw  = (const float*)d_in[10];
  const float* dtb  = (const float*)d_in[11];
  const float* Alog = (const float*)d_in[12];
  const float* Dp   = (const float*)d_in[13];
  const float* ow   = (const float*)d_in[14];
  const float* nfw  = (const float*)d_in[15];
  const float* hw   = (const float*)d_in[16];
  const float* hb   = (const float*)d_in[17];
  float* out = (float*)d_out;

  char* base = (char*)d_ws;
  size_t off = 0;
  auto take = [&](size_t bytes) -> char* {
    off = (off + 255) & ~(size_t)255;
    char* p = base + off;
    off += bytes;
    return p;
  };
  _Float16* w_in_h  = (_Float16*)take((size_t)DEPTH * 2 * D_INNER * D_MODEL * 2);
  _Float16* w_x_h   = (_Float16*)take((size_t)DEPTH * XPROJ_OUT * D_INNER * 2);
  _Float16* w_out_h = (_Float16*)take((size_t)DEPTH * D_MODEL * D_INNER * 2);
  float*    hidden   = (float*)take((size_t)NTOK * D_MODEL * 4);
  float*    residual = (float*)take((size_t)NTOK * D_MODEL * 4);
  _Float16* xnorm_h  = (_Float16*)take((size_t)NTOK * D_MODEL * 2);
  float*    xz       = (float*)take((size_t)NTOK * 2 * D_INNER * 4);
  float*    xc       = (float*)take((size_t)NTOK * D_INNER * 4);
  _Float16* xch      = (_Float16*)take((size_t)NTOK * D_INNER * 2);
  float*    xdbl     = (float*)take((size_t)NTOK * XPROJ_OUT * 4);
  float*    dtf      = (float*)take((size_t)NTOK * D_INNER * 4);
  _Float16* ygh      = (_Float16*)take((size_t)NTOK * D_INNER * 2);

  // weight conversion
  {
    int n1 = DEPTH * 2 * D_INNER * D_MODEL;
    k_f32_to_f16<<<(n1 + 255) / 256, 256, 0, stream>>>(inw, w_in_h, n1);
    int n2 = DEPTH * XPROJ_OUT * D_INNER;
    k_f32_to_f16<<<(n2 + 255) / 256, 256, 0, stream>>>(xw, w_x_h, n2);
    int n3 = DEPTH * D_MODEL * D_INNER;
    k_f32_to_f16<<<(n3 + 255) / 256, 256, 0, stream>>>(ow, w_out_h, n3);
  }

  k_patch_embed<<<dim3(LSEQ, BATCH), D_MODEL, 0, stream>>>(x, pw, pb, ct, pe, hidden, residual);

  const int tilesM2 = (NTOK + 31) / 32;  // 50 (32-row blocks, edge masked)
  const int tilesM1 = (NTOK + 15) / 16;  // 99
  for (int L = 0; L < DEPTH; ++L) {
    k_prenorm<<<NTOK, D_MODEL, 0, stream>>>(hidden, residual, nw + L * D_MODEL, xnorm_h);
    {
      int tN2 = (2 * D_INNER) / 32;  // 24
      k_gemm_f16_2x2<D_MODEL><<<tilesM2 * tN2, 32, 0, stream>>>(
          xnorm_h, w_in_h + (size_t)L * 2 * D_INNER * D_MODEL, xz,
          NTOK, 2 * D_INNER, 2 * D_INNER, tN2);
    }
    {
      int n = NTOK * D_INNER;
      k_conv_silu<<<(n + 255) / 256, 256, 0, stream>>>(
          xz, cw + (size_t)L * D_INNER * D_CONVK, cb + L * D_INNER, xc, xch);
    }
    {
      int tN = (XPROJ_OUT + 15) / 16;  // 3 (N padded; stores masked to 44)
      k_gemm_f16<D_INNER><<<tilesM1 * tN, 32, 0, stream>>>(
          xch, w_x_h + (size_t)L * XPROJ_OUT * D_INNER, xdbl,
          NTOK, XPROJ_OUT, XPROJ_OUT, tN);
    }
    {
      int n = NTOK * D_INNER;
      k_dt_softplus<<<(n + 255) / 256, 256, 0, stream>>>(
          xdbl, dtw + (size_t)L * D_INNER * DT_RANK, dtb + L * D_INNER, dtf);
    }
    k_scan_gate<<<dim3(3, BATCH), 128, 0, stream>>>(
        dtf, xc, xdbl, xz, Alog + (size_t)L * D_INNER * D_STATE, Dp + L * D_INNER, ygh);
    {
      int tN2 = D_MODEL / 32;  // 6
      k_gemm_f16_2x2<D_INNER><<<tilesM2 * tN2, 32, 0, stream>>>(
          ygh, w_out_h + (size_t)L * D_MODEL * D_INNER, hidden,
          NTOK, D_MODEL, D_MODEL, tN2);
    }
  }

  k_head<<<BATCH, 256, 0, stream>>>(hidden, residual, nfw, hw, hb, out);
}
